// MoEParser_7876970021458
// MI455X (gfx1250) — matrix operand order repaired and checked by reading, not compiled, not run on hardware
//
#include <hip/hip_runtime.h>
#include <hip/hip_bf16.h>
#include <math.h>

typedef __attribute__((ext_vector_type(16))) _Float16 v16h;
typedef __attribute__((ext_vector_type(8)))  _Float16 v8h;
typedef __attribute__((ext_vector_type(8)))  float    v8f;

#define EPSC 1e-5f
#define MAXN (1.0f - 1e-4f)

__device__ __forceinline__ float artanhf_(float x) {
    x = fminf(fmaxf(x, -1.0f + 1e-5f), 1.0f - 1e-5f);
    return 0.5f * (log1pf(x) - log1pf(-x));
}
__device__ __forceinline__ float wave_sum(float v) {
#pragma unroll
    for (int m = 16; m >= 1; m >>= 1) v += __shfl_xor(v, m, 32);
    return v;
}
__device__ __forceinline__ float half_sum(float v) {   // reduce within each 16-lane half
#pragma unroll
    for (int m = 8; m >= 1; m >>= 1) v += __shfl_xor(v, m, 32);
    return v;
}

// ---- WMMA fragment loaders (row-major, K contiguous; C = A(MxK) * B(NxK)^T) ----
// A 16x32 f16: lane l (half h=l>>4, r=l&15): elems 0..15 = K[16h..16h+15]
__device__ __forceinline__ v16h load_fragA(const _Float16* base, int row0, int ld) {
    const int lane = threadIdx.x & 31;
    const int h = lane >> 4, r = lane & 15;
    const _Float16* p = base + (size_t)(row0 + r) * ld + 16 * h;
    v16h f;
    *(v8h*)&f       = *(const v8h*)(p);        // 16B
    *((v8h*)&f + 1) = *(const v8h*)(p + 8);    // 16B
    return f;
}
// B 32x16 f16 (stored as N rows x K cols): lane l = col N, elems 0..15 = K[16h..16h+15]
__device__ __forceinline__ v16h load_fragB(const _Float16* base, int row0, int ld) {
    const int lane = threadIdx.x & 31;
    const int h = lane >> 4, r = lane & 15;
    return *(const v16h*)(base + (size_t)(row0 + r) * ld + 16 * h);  // 32B
}
__device__ __forceinline__ v8f wmma_f16(v16h a, v16h b, v8f c) {
    return __builtin_amdgcn_wmma_f32_16x16x32_f16(false, a, false, b, (short)0, c, false, false);
}

// ---------------- prep: f16 casts, rel transpose, bias projection ----------------
__global__ void k_prep(const float* exp_W, const float* rel_T, const float* rel_S,
                       _Float16* Wf16, _Float16* rel16) {
    int i = blockIdx.x * blockDim.x + threadIdx.x;
    if (i < 17 * 64 * 64) Wf16[i] = (_Float16)exp_W[i];
    if (i < 2 * 40 * 32 * 32) {
        int w = i / (40 * 32 * 32);
        int rem = i % (40 * 32 * 32);
        int n = rem >> 5, d = rem & 31;          // n = r*32+e
        int r = n >> 5, e = n & 31;
        const float* src = w ? rel_S : rel_T;    // rel[r,d,e] -> rel16[(w*1280+n)*32+d]
        rel16[i] = (_Float16)src[((r << 5) + d) * 32 + e];
    }
}
__global__ void k_prep_bias(const float* exp_bias, float* bp, float* y2) {  // 17 waves
    int w = threadIdx.x >> 5, lane = threadIdx.x & 31;
    float2 v = *(const float2*)(exp_bias + w * 64 + lane * 2);
    float s = wave_sum(v.x * v.x + v.y * v.y);
    float n = sqrtf(fmaxf(s, EPSC * EPSC));
    float f = fminf(1.0f, MAXN / n);
    bp[w * 64 + lane * 2] = f * v.x;
    bp[w * 64 + lane * 2 + 1] = f * v.y;
    if (lane == 0) y2[w] = f * f * s;
}

// ---------------- embed + conv + router + tangent round-trip (1 wave / token) ----------------
__global__ void k_embed_router(const int* tokens, const float* emb,
                               const float* conv_w, const float* conv_b,
                               const float* lin_w, const float* lin_b,
                               float* logp_out, float* p_ws, _Float16* x16, float* xn_ws) {
    const int bt = (blockIdx.x * blockDim.x + threadIdx.x) >> 5;  // token id, grid exact = 4096
    const int lane = threadIdx.x & 31;
    const int t = bt & 511;
    const int d0 = lane * 2;

    int tok = tokens[bt];
    float2 h = *(const float2*)(emb + (size_t)tok * 64 + d0);
    float2 hm = make_float2(0.f, 0.f), hp = make_float2(0.f, 0.f);
    if (t > 0)   { int tk = tokens[bt - 1]; hm = *(const float2*)(emb + (size_t)tk * 64 + d0); }
    if (t < 511) { int tk = tokens[bt + 1]; hp = *(const float2*)(emb + (size_t)tk * 64 + d0); }

    float hc0 = hm.x * conv_w[d0 * 3] + h.x * conv_w[d0 * 3 + 1] + hp.x * conv_w[d0 * 3 + 2] + conv_b[d0];
    float hc1 = hm.y * conv_w[(d0 + 1) * 3] + h.y * conv_w[(d0 + 1) * 3 + 1] + hp.y * conv_w[(d0 + 1) * 3 + 2] + conv_b[d0 + 1];

    float lg[17];
#pragma unroll
    for (int k = 0; k < 17; ++k) {
        float v = hc0 * lin_w[d0 * 17 + k] + hc1 * lin_w[(d0 + 1) * 17 + k];
        lg[k] = wave_sum(v) + lin_b[k];
    }
    float mv = -3.4e38f;
#pragma unroll
    for (int k = 0; k < 17; ++k) mv = fmaxf(mv, lg[k]);
    float se = 0.f;
#pragma unroll
    for (int k = 0; k < 17; ++k) se += expf(lg[k] - mv);
    float lz = mv + logf(se);
    if (lane < 17) {
        float lp = lg[lane] - lz;
        logp_out[bt * 17 + lane] = lp;
        p_ws[bt * 17 + lane] = expf(lp);
    }

    // x = expmap0(logmap0(project(h0)))
    float s0 = wave_sum(h.x * h.x + h.y * h.y);
    float n = sqrtf(fmaxf(s0, EPSC * EPSC));
    float f = fminf(1.0f, MAXN / n);
    float n1 = sqrtf(fmaxf(f * f * s0, EPSC * EPSC));
    float art = artanhf_(n1);
    float lv2 = (art / n1) * (art / n1) * f * f * s0;
    float n2 = sqrtf(fmaxf(lv2, EPSC * EPSC));
    float c = f * (art / n1) * (tanhf(n2) / n2);
    x16[(size_t)bt * 64 + d0] = (_Float16)(c * h.x);
    x16[(size_t)bt * 64 + d0 + 1] = (_Float16)(c * h.y);
    if (lane == 0) xn_ws[bt] = sqrtf(fmaxf(c * c * s0, EPSC * EPSC));
}

// ---------------- GEMM: mx[4096,1088] = x16[4096,64] @ Wf16[1088,64]^T ----------------
__global__ void k_gemm_mx(const _Float16* x16, const _Float16* Wf16, float* mx) {
    const int w = threadIdx.x >> 5, lane = threadIdx.x & 31;
    const int h = lane >> 4, r = lane & 15;
    const int m0 = blockIdx.x * 64 + w * 16;
    const int n0 = blockIdx.y * 64;
    v16h a0 = load_fragA(x16, m0, 64);       // K 0..31
    v16h a1 = load_fragA(x16 + 32, m0, 64);  // K 32..63
#pragma unroll
    for (int nt = 0; nt < 4; ++nt) {
        int nb = n0 + nt * 16;
        v16h b0 = load_fragB(Wf16, nb, 64);
        v16h b1 = load_fragB(Wf16 + 32, nb, 64);
        v8f c = {};
        c = wmma_f16(a0, b0, c);
        c = wmma_f16(a1, b1, c);
#pragma unroll
        for (int v = 0; v < 8; ++v)
            mx[(size_t)(m0 + v + 8 * h) * 1088 + (nb + r)] = c[v];
    }
}

// ---------------- 17 hyperbolic experts fused -> y_avg (1 wave / token) ----------------
__global__ void k_experts(const float* mx, const float* xn_ws, const float* p_ws,
                          const float* bp, const float* y2a,
                          const float* ln_g, const float* ln_bt, _Float16* yavg16) {
    const int m = (blockIdx.x * blockDim.x + threadIdx.x) >> 5;  // grid exact = 4096
    const int lane = threadIdx.x & 31;
    const int e0 = (lane & 15) * 2;

    float xn = xn_ws[m];
    float art_xn = artanhf_(xn);
    float acc0 = 0.f, acc1 = 0.f;
    for (int k = 0; k < 17; ++k) {
        float2 v = *(const float2*)(mx + (size_t)m * 1088 + k * 64 + lane * 2);
        float mn2 = wave_sum(v.x * v.x + v.y * v.y);
        float mn = sqrtf(fmaxf(mn2, EPSC * EPSC));
        float tt = tanhf(mn / xn * art_xn);
        float s = tt / mn;
        float mv0 = s * v.x, mv1 = s * v.y;
        float x2 = s * s * mn2;
        float b0 = bp[k * 64 + lane * 2], b1 = bp[k * 64 + lane * 2 + 1];
        float xy = wave_sum(mv0 * b0 + mv1 * b1);
        float y2 = y2a[k];
        float den = fmaxf(1.f + 2.f * xy + x2 * y2, EPSC);
        float ca = (1.f + 2.f * xy + y2) / den;
        float cb = (1.f - x2) / den;
        float yv0 = ca * mv0 + cb * b0, yv1 = ca * mv1 + cb * b1;
        float s2 = wave_sum(yv0 * yv0 + yv1 * yv1);
        float n = sqrtf(fmaxf(s2, EPSC * EPSC));
        float f = fminf(1.f, MAXN / n);
        yv0 *= f; yv1 *= f;
        float n1 = sqrtf(fmaxf(f * f * s2, EPSC * EPSC));
        float lt = artanhf_(n1) / n1;
        float yt0 = lt * yv0, yt1 = lt * yv1;
        // GLU across halves: lanes 0-15 hold a, 16-31 hold g for the same e
        float o0 = __shfl_xor(yt0, 16, 32);
        float o1 = __shfl_xor(yt1, 16, 32);
        float a0 = (lane < 16) ? yt0 : o0;
        float a1 = (lane < 16) ? yt1 : o1;
        float g0 = (lane < 16) ? o0 : yt0;
        float g1 = (lane < 16) ? o1 : yt1;
        float u0 = a0 / (1.f + expf(-g0));
        float u1 = a1 / (1.f + expf(-g1));
        // LayerNorm over E=32 (both halves hold identical copies)
        float mu = half_sum(u0 + u1) * (1.f / 32.f);
        float var = half_sum(u0 * u0 + u1 * u1) * (1.f / 32.f) - mu * mu;
        float rstd = rsqrtf(var + 1e-5f);
        float l0 = (u0 - mu) * rstd * ln_g[k * 32 + e0] + ln_bt[k * 32 + e0];
        float l1 = (u1 - mu) * rstd * ln_g[k * 32 + e0 + 1] + ln_bt[k * 32 + e0 + 1];
        float n3 = sqrtf(fmaxf(half_sum(l0 * l0 + l1 * l1), EPSC * EPSC));
        float ec = tanhf(n3) / n3;
        float pk = p_ws[m * 17 + k];
        acc0 += pk * ec * l0;
        acc1 += pk * ec * l1;
    }
    if (lane < 16) {
        yavg16[(size_t)m * 32 + e0] = (_Float16)acc0;
        yavg16[(size_t)m * 32 + e0 + 1] = (_Float16)acc1;
    }
}

// ---------------- GEMM: [Th|Sd] = y_avg[4096,32] @ rel16[2560,32]^T, scatter to [b,r,t,e] ----------------
__global__ void k_gemm_rel(const _Float16* yavg16, const _Float16* rel16,
                           _Float16* Th16, _Float16* Sd16) {
    const int w = threadIdx.x >> 5, lane = threadIdx.x & 31;
    const int h = lane >> 4, r = lane & 15;
    const int m0 = blockIdx.x * 64 + w * 16;
    const int n0 = blockIdx.y * 64;
    v16h a = load_fragA(yavg16, m0, 32);
#pragma unroll
    for (int nt = 0; nt < 4; ++nt) {
        int nb = n0 + nt * 16;
        v16h bf = load_fragB(rel16, nb, 32);
        v8f c = {};
        c = wmma_f16(a, bf, c);
        int ncol = nb + r;
        int which = ncol >= 1280;
        int nn = which ? (ncol - 1280) : ncol;
        int rr = nn >> 5, e = nn & 31;
        _Float16* dst = which ? Sd16 : Th16;
#pragma unroll
        for (int v = 0; v < 8; ++v) {
            int mrow = m0 + v + 8 * h;
            int bb = mrow >> 9, t = mrow & 511;
            dst[(((size_t)(bb * 40 + rr) * 512 + t) << 5) + e] = (_Float16)c[v];
        }
    }
}

// ---------------- biaffine scores fused with max over 40 relations + diag mask ----------------
__global__ void k_scores(const _Float16* Th16, const _Float16* Sd16, float* score) {
    const int w = threadIdx.x >> 5, lane = threadIdx.x & 31;
    const int h = lane >> 4, r = lane & 15;
    const int b = blockIdx.z, it = blockIdx.y, jt = blockIdx.x;
    const int i0 = it * 64 + w * 16, j0 = jt * 64;
    float acc[4][8];
#pragma unroll
    for (int nt = 0; nt < 4; ++nt)
#pragma unroll
        for (int v = 0; v < 8; ++v) acc[nt][v] = -3.4e38f;

    for (int rr = 0; rr < 40; ++rr) {
        const _Float16* tb = Th16 + (size_t)(b * 40 + rr) * 512 * 32;
        const _Float16* sb = Sd16 + (size_t)(b * 40 + rr) * 512 * 32;
        if (rr + 1 < 40) {
            __builtin_prefetch(tb + 512 * 32 + (size_t)(i0 + r) * 32, 0, 0);
            __builtin_prefetch(sb + 512 * 32 + (size_t)(j0 + r * 4) * 32, 0, 0);
        }
        v16h a = load_fragA(tb, i0, 32);
#pragma unroll
        for (int nt = 0; nt < 4; ++nt) {
            v16h bf = load_fragB(sb, j0 + nt * 16, 32);
            v8f c = {};
            c = wmma_f16(a, bf, c);
#pragma unroll
            for (int v = 0; v < 8; ++v) acc[nt][v] = fmaxf(acc[nt][v], c[v] * (1.f / 32.f));
        }
    }
    float* so = score + (size_t)b * 512 * 512;
#pragma unroll
    for (int nt = 0; nt < 4; ++nt)
#pragma unroll
        for (int v = 0; v < 8; ++v) {
            int i = i0 + v + 8 * h;
            int j = j0 + nt * 16 + r;
            so[(size_t)i * 512 + j] = (i == j) ? -1e9f : acc[nt][v];
        }
}

// ---------------- greedy head selection (first-max argmax over axis=1) ----------------
__global__ void k_tree(const float* score, float* tree) {
    int b = blockIdx.x, j = threadIdx.x;
    const float* s = score + (size_t)b * 512 * 512 + j;
    float best = s[0];
    int arg = 0;
    for (int i = 1; i < 512; ++i) {
        float v = s[(size_t)i * 512];
        if (v > best) { best = v; arg = i; }
    }
    tree[b * 512 + j] = (float)arg;
}

extern "C" void kernel_launch(void* const* d_in, const int* in_sizes, int n_in,
                              void* d_out, int out_size, void* d_ws, size_t ws_size,
                              hipStream_t stream) {
    (void)in_sizes; (void)n_in; (void)out_size; (void)ws_size;
    const int*   tokens   = (const int*)d_in[0];
    const float* emb      = (const float*)d_in[1];
    const float* conv_w   = (const float*)d_in[2];
    const float* conv_b   = (const float*)d_in[3];
    const float* lin_w    = (const float*)d_in[4];
    const float* lin_b    = (const float*)d_in[5];
    const float* exp_W    = (const float*)d_in[6];
    const float* exp_bias = (const float*)d_in[7];
    const float* ln_g     = (const float*)d_in[8];
    const float* ln_bt    = (const float*)d_in[9];
    const float* rel_T    = (const float*)d_in[10];
    const float* rel_S    = (const float*)d_in[11];

    float* out       = (float*)d_out;
    float* tree_out  = out;                       // [8,512] (as float)
    float* score_out = out + 4096;                // [8,512,512]
    float* logp_out  = out + 4096 + 8 * 512 * 512;// [8,512,17]

    char* ws = (char*)d_ws;
    size_t off = 0;
    auto carve = [&](size_t bytes) -> char* {
        char* p = ws + off;
        off += (bytes + 255) & ~(size_t)255;
        return p;
    };
    _Float16* x16    = (_Float16*)carve((size_t)4096 * 64 * 2);
    float*    xn_ws  = (float*)   carve((size_t)4096 * 4);
    float*    p_ws   = (float*)   carve((size_t)4096 * 17 * 4);
    _Float16* Wf16   = (_Float16*)carve((size_t)17 * 64 * 64 * 2);
    float*    bp     = (float*)   carve((size_t)17 * 64 * 4);
    float*    y2     = (float*)   carve(256);
    _Float16* rel16  = (_Float16*)carve((size_t)2560 * 32 * 2);
    float*    mx     = (float*)   carve((size_t)4096 * 1088 * 4);
    _Float16* yavg16 = (_Float16*)carve((size_t)4096 * 32 * 2);
    _Float16* Th16   = (_Float16*)carve((size_t)8 * 40 * 512 * 32 * 2);
    _Float16* Sd16   = (_Float16*)carve((size_t)8 * 40 * 512 * 32 * 2);

    k_prep<<<320, 256, 0, stream>>>(exp_W, rel_T, rel_S, Wf16, rel16);
    k_prep_bias<<<1, 544, 0, stream>>>(exp_bias, bp, y2);
    k_embed_router<<<1024, 128, 0, stream>>>(tokens, emb, conv_w, conv_b, lin_w, lin_b,
                                             logp_out, p_ws, x16, xn_ws);
    k_gemm_mx<<<dim3(64, 17), 128, 0, stream>>>(x16, Wf16, mx);
    k_experts<<<1024, 128, 0, stream>>>(mx, xn_ws, p_ws, bp, y2, ln_g, ln_bt, yavg16);
    k_gemm_rel<<<dim3(64, 40), 128, 0, stream>>>(yavg16, rel16, Th16, Sd16);
    k_scores<<<dim3(8, 8, 8), 128, 0, stream>>>(Th16, Sd16, score_out);
    k_tree<<<8, 512, 0, stream>>>(score_out, tree_out);
}
